// MoS_EXT_44306882626206
// MI455X (gfx1250) — matrix operand order repaired and checked by reading, not compiled, not run on hardware
//
#include <hip/hip_runtime.h>
#include <hip/hip_bf16.h>

typedef __attribute__((ext_vector_type(16))) __bf16 v16bf;
typedef __attribute__((ext_vector_type(8)))  __bf16 v8bf;
typedef __attribute__((ext_vector_type(8)))  float  v8f;

#define VOCAB 32000
#define IDIM  512
#define HDIM  1024
#define IMG   2048
#define BB    64
#define TT    64
#define NROWS (BB*(TT-1))   // 4032

// ---------------- WMMA fragment loaders (16-bit, 16x16x32) ----------------
// A (16x32, row-major source): lane L<16 -> row L, K {kb..kb+7, kb+16..kb+23}, kb = (L>=16)*8
__device__ inline v16bf frag_a16(const __bf16* base, size_t row_stride, int m0, int k0, int lane) {
    int r  = lane & 15;
    int kb = (lane >> 4) << 3;                 // 0 or 8
    const __bf16* p = base + (size_t)(m0 + r) * row_stride + k0 + kb;
    union { v16bf v; v8bf h[2]; } f;
    f.h[0] = *(const v8bf*)(p);
    f.h[1] = *(const v8bf*)(p + 16);
    return f.v;
}
// B (32x16): column n = weight row (n0 + L&15); lane holds K {kb..kb+7, kb+16..kb+23}, kb = (L>=16)*8
__device__ inline v16bf frag_b16(const __bf16* wbase, size_t row_stride, int n0, int k0, int lane) {
    int c  = lane & 15;
    int kb = (lane >> 4) << 3;                 // 0 or 8
    const __bf16* p = wbase + (size_t)(n0 + c) * row_stride + k0 + kb;
    union { v16bf v; v8bf h[2]; } f;
    f.h[0] = *(const v8bf*)(p);
    f.h[1] = *(const v8bf*)(p + 16);
    return f.v;
}

__device__ inline float sigm(float x) { return 1.0f / (1.0f + __expf(-x)); }

#define WMMA_BF16(a, b, c) \
    __builtin_amdgcn_wmma_f32_16x16x32_bf16(false, (a), false, (b), (short)0, (c), false, false)

struct FragSet { v16bf a, b0, b1, b2, b3; };

__device__ inline FragSet load_set(const __bf16* abase, size_t astride, int m0,
                                   const __bf16* wbase, size_t bstride, int n0,
                                   int k0, int lane) {
    FragSet f;
    f.a  = frag_a16(abase, astride, m0, k0, lane);
    f.b0 = frag_b16(wbase, bstride, n0,      k0, lane);
    f.b1 = frag_b16(wbase, bstride, n0 + 16, k0, lane);
    f.b2 = frag_b16(wbase, bstride, n0 + 32, k0, lane);
    f.b3 = frag_b16(wbase, bstride, n0 + 48, k0, lane);
    return f;
}

__device__ inline void wmma4(const FragSet& f, v8f& a0, v8f& a1, v8f& a2, v8f& a3) {
    a0 = WMMA_BF16(f.a, f.b0, a0);
    a1 = WMMA_BF16(f.a, f.b1, a1);
    a2 = WMMA_BF16(f.a, f.b2, a2);
    a3 = WMMA_BF16(f.a, f.b3, a3);
}

// Double-buffered 16x64-strip GEMM over K (multiple of 64).
// NOTE: issues one extra prefetch fragment-set at k==K (over-reads <=64B per row);
// all source buffers have live allocations after them (or LDS tail padding), and the
// garbage set is never fed to a WMMA.
__device__ inline void gemm_pipelined(const __bf16* abase, size_t astride, int m0,
                                      const __bf16* wbase, size_t bstride, int n0,
                                      int K, int lane,
                                      v8f& a0, v8f& a1, v8f& a2, v8f& a3) {
    FragSet s0 = load_set(abase, astride, m0, wbase, bstride, n0, 0, lane);
    for (int k = 0; k < K; k += 64) {
        FragSet s1 = load_set(abase, astride, m0, wbase, bstride, n0, k + 32, lane);
        __builtin_amdgcn_sched_barrier(0);
        wmma4(s0, a0, a1, a2, a3);
        __builtin_amdgcn_sched_barrier(0);
        s0 = load_set(abase, astride, m0, wbase, bstride, n0, k + 64, lane);
        __builtin_amdgcn_sched_barrier(0);
        wmma4(s1, a0, a1, a2, a3);
        __builtin_amdgcn_sched_barrier(0);
    }
}

// ---------------- small utility kernels ----------------
__global__ void zero_acc_kernel(float* acc) { acc[0] = 0.f; acc[1] = 0.f; }

__global__ void convert_bf16_kernel(const float* __restrict__ in, __bf16* __restrict__ out, int n) {
    int i = blockIdx.x * blockDim.x + threadIdx.x;
    if (i < n) out[i] = (__bf16)in[i];
}

// x[b][t][k] = bf16(embed[tgt_ids[b][t]][k]),  t in [0,62]
__global__ void gather_x_kernel(const float* __restrict__ embed, const int* __restrict__ tgt_ids,
                                __bf16* __restrict__ x_bf, int n) {
    int i = blockIdx.x * blockDim.x + threadIdx.x;
    if (i >= n) return;
    int k   = i & (IDIM - 1);
    int bt  = i / IDIM;
    int t   = bt % (TT - 1);
    int b   = bt / (TT - 1);
    int tok = tgt_ids[b * TT + t];
    x_bf[i] = (__bf16)embed[(size_t)tok * IDIM + k];
}

// ---------------- init: [h0|c0] = tanh(src @ [W_h;W_c]^T + b), 16x64 strip/wave ----------------
__global__ __launch_bounds__(32) void init_hc_kernel(const __bf16* __restrict__ src_bf,
                                                     const __bf16* __restrict__ whc_bf,
                                                     const float* __restrict__ b_h,
                                                     const float* __restrict__ b_c,
                                                     float* __restrict__ c_state,
                                                     __bf16* __restrict__ h_bf) {
    int bid = blockIdx.x;                 // 4 row tiles x 32 col groups (64 cols each)
    int n0  = (bid & 31) * 64;
    int m0  = (bid >> 5) * 16;
    int lane = threadIdx.x & 31;
    v8f acc0 = {0,0,0,0,0,0,0,0}, acc1 = acc0, acc2 = acc0, acc3 = acc0;
    gemm_pipelined(src_bf, IMG, m0, whc_bf, IMG, n0, IMG, lane, acc0, acc1, acc2, acc3);
    int half = lane >> 4, c = lane & 15;
    v8f accs[4] = {acc0, acc1, acc2, acc3};
    #pragma unroll
    for (int tn = 0; tn < 4; ++tn) {
        int col = n0 + tn * 16 + c;
        float bias = (col < HDIM) ? b_h[col] : b_c[col - HDIM];
        #pragma unroll
        for (int j = 0; j < 8; ++j) {
            int row = m0 + j + 8 * half;
            float v = tanhf(accs[tn][j] + bias);
            if (col < HDIM) h_bf[(size_t)row * HDIM + col] = (__bf16)v;
            else            c_state[(size_t)row * HDIM + (col - HDIM)] = v;
        }
    }
}

// ---------------- LSTM step part 1: G = x_t @ W_ih^T + h @ W_hh^T, 16x64 strip/wave ----------------
__global__ __launch_bounds__(32) void gates_gemm_kernel(const __bf16* __restrict__ x_bf,
                                                        const __bf16* __restrict__ h_bf,
                                                        const __bf16* __restrict__ wih,
                                                        const __bf16* __restrict__ whh,
                                                        float* __restrict__ G, int t) {
    int bid = blockIdx.x;                 // 4 row tiles x 64 col groups
    int n0  = (bid & 63) * 64;
    int m0  = (bid >> 6) * 16;
    int lane = threadIdx.x & 31;
    v8f acc0 = {0,0,0,0,0,0,0,0}, acc1 = acc0, acc2 = acc0, acc3 = acc0;
    const __bf16* xbase = x_bf + (size_t)t * IDIM;   // row b at xbase + b*(63*IDIM)
    gemm_pipelined(xbase, (size_t)(TT - 1) * IDIM, m0, wih, IDIM, n0, IDIM, lane,
                   acc0, acc1, acc2, acc3);
    gemm_pipelined(h_bf, HDIM, m0, whh, HDIM, n0, HDIM, lane,
                   acc0, acc1, acc2, acc3);
    int half = lane >> 4, c = lane & 15;
    v8f accs[4] = {acc0, acc1, acc2, acc3};
    #pragma unroll
    for (int tn = 0; tn < 4; ++tn)
        #pragma unroll
        for (int j = 0; j < 8; ++j)
            G[(size_t)(m0 + j + 8 * half) * (4 * HDIM) + n0 + tn * 16 + c] = accs[tn][j];
}

// ---------------- LSTM step part 2: gates -> c,h; emit bf16 h and outs row ----------------
__global__ __launch_bounds__(256) void gate_update_kernel(const float* __restrict__ G,
                                                          const float* __restrict__ b_ih,
                                                          const float* __restrict__ b_hh,
                                                          float* __restrict__ c_state,
                                                          __bf16* __restrict__ h_bf,
                                                          __bf16* __restrict__ outs_bf, int t) {
    int i = blockIdx.x * blockDim.x + threadIdx.x;
    if (i >= BB * HDIM) return;
    int b = i >> 10, u = i & (HDIM - 1);
    size_t base = (size_t)b * (4 * HDIM);
    float iv = G[base + u]            + b_ih[u]            + b_hh[u];
    float fv = G[base + HDIM + u]     + b_ih[HDIM + u]     + b_hh[HDIM + u];
    float gv = G[base + 2*HDIM + u]   + b_ih[2*HDIM + u]   + b_hh[2*HDIM + u];
    float ov = G[base + 3*HDIM + u]   + b_ih[3*HDIM + u]   + b_hh[3*HDIM + u];
    float cn = sigm(fv) * c_state[i] + sigm(iv) * tanhf(gv);
    float hn = sigm(ov) * tanhf(cn);
    c_state[i] = cn;
    __bf16 hb = (__bf16)hn;
    h_bf[i] = hb;
    outs_bf[((size_t)b * (TT - 1) + t) * HDIM + u] = hb;
}

// ---------------- streaming log-sum-exp of logits = outs @ W_out^T + b_out ----------------
// Each block: 16 rows (A tile in LDS), 8 waves strip-mine 500 groups of 64 vocab columns.
__global__ __launch_bounds__(256) void lse_kernel(const __bf16* __restrict__ outs_bf,
                                                  const __bf16* __restrict__ wout_bf,
                                                  const float* __restrict__ b_out,
                                                  float* __restrict__ lse) {
    __shared__ __align__(16) __bf16 Atile[16 * HDIM + 64];  // +64: pipeline over-read tail
    __shared__ float red_m[8][16];
    __shared__ float red_s[8][16];
    int r0  = blockIdx.x * 16;                          // 252 row tiles
    int tid = threadIdx.x;
    // cooperative 16B-wide copy of A tile (16 rows x 1024 K, bf16)
    for (int i = tid; i < (16 * HDIM) / 8; i += 256) {
        int row = i >> 7;
        int k8  = (i & 127) << 3;
        *(uint4*)&Atile[row * HDIM + k8] = *(const uint4*)&outs_bf[(size_t)(r0 + row) * HDIM + k8];
    }
    __syncthreads();
    int wave = tid >> 5, lane = tid & 31;
    int half = lane >> 4, c = lane & 15;
    float m[8], s[8];
    #pragma unroll
    for (int j = 0; j < 8; ++j) { m[j] = -3.4e38f; s[j] = 0.f; }
    for (int g = wave; g < VOCAB / 64; g += 8) {        // 500 groups of 64 columns
        int n0 = g * 64;
        v8f acc0 = {0,0,0,0,0,0,0,0}, acc1 = acc0, acc2 = acc0, acc3 = acc0;
        gemm_pipelined(Atile, HDIM, 0, wout_bf, HDIM, n0, HDIM, lane,
                       acc0, acc1, acc2, acc3);
        float bias0 = b_out[n0 + c];
        float bias1 = b_out[n0 + 16 + c];
        float bias2 = b_out[n0 + 32 + c];
        float bias3 = b_out[n0 + 48 + c];
        #pragma unroll
        for (int j = 0; j < 8; ++j) {
            float v0 = acc0[j] + bias0;
            float v1 = acc1[j] + bias1;
            float v2 = acc2[j] + bias2;
            float v3 = acc3[j] + bias3;
            float tm = fmaxf(fmaxf(v0, v1), fmaxf(v2, v3));   // row-max over 64 cols
            tm = fmaxf(tm, __shfl_xor(tm, 1, 32));
            tm = fmaxf(tm, __shfl_xor(tm, 2, 32));
            tm = fmaxf(tm, __shfl_xor(tm, 4, 32));
            tm = fmaxf(tm, __shfl_xor(tm, 8, 32));
            float nm = fmaxf(m[j], tm);
            float e = __expf(v0 - nm) + __expf(v1 - nm) + __expf(v2 - nm) + __expf(v3 - nm);
            e += __shfl_xor(e, 1, 32);
            e += __shfl_xor(e, 2, 32);
            e += __shfl_xor(e, 4, 32);
            e += __shfl_xor(e, 8, 32);
            s[j] = s[j] * __expf(m[j] - nm) + e;
            m[j] = nm;
        }
    }
    if (c == 0) {
        #pragma unroll
        for (int j = 0; j < 8; ++j) { red_m[wave][j + 8 * half] = m[j]; red_s[wave][j + 8 * half] = s[j]; }
    }
    __syncthreads();
    if (tid < 16) {
        float M = -3.4e38f;
        for (int w = 0; w < 8; ++w) M = fmaxf(M, red_m[w][tid]);
        float S = 0.f;
        for (int w = 0; w < 8; ++w) S += red_s[w][tid] * __expf(red_m[w][tid] - M);
        lse[r0 + tid] = M + __logf(S);
    }
}

// ---------------- masked NLL: target logit dot + accumulate ----------------
__global__ __launch_bounds__(256) void loss_kernel(const __bf16* __restrict__ outs_bf,
                                                   const float* __restrict__ W_out,
                                                   const float* __restrict__ b_out,
                                                   const float* __restrict__ lse,
                                                   const int* __restrict__ tgt_ids,
                                                   float* __restrict__ acc) {
    int wave = threadIdx.x >> 5, lane = threadIdx.x & 31;
    int row = blockIdx.x * 8 + wave;
    if (row >= NROWS) return;
    int b = row / (TT - 1), t = row % (TT - 1);
    int tok = tgt_ids[b * TT + t + 1];
    if (tok == 0) return;                                  // mask
    const __bf16* hrow = outs_bf + (size_t)row * HDIM;
    const float*  wrow = W_out + (size_t)tok * HDIM;
    float sum = 0.f;
    for (int k = lane; k < HDIM; k += 32) sum += (float)hrow[k] * wrow[k];
    sum += __shfl_xor(sum, 16, 32);
    sum += __shfl_xor(sum, 8, 32);
    sum += __shfl_xor(sum, 4, 32);
    sum += __shfl_xor(sum, 2, 32);
    sum += __shfl_xor(sum, 1, 32);
    if (lane == 0) {
        float tok_lp = sum + b_out[tok] - lse[row];
        atomicAdd(&acc[0], -tok_lp);
        atomicAdd(&acc[1], 1.f);
    }
}

__global__ void finalize_kernel(const float* __restrict__ acc, float* __restrict__ out) {
    out[0] = acc[0] / fmaxf(acc[1], 1.f);
    out[1] = acc[1];
}

// ---------------- host-side orchestration ----------------
extern "C" void kernel_launch(void* const* d_in, const int* in_sizes, int n_in,
                              void* d_out, int out_size, void* d_ws, size_t ws_size,
                              hipStream_t stream) {
    const float* src     = (const float*)d_in[0];
    const int*   tgt_ids = (const int*)  d_in[1];
    const float* embed   = (const float*)d_in[2];
    const float* W_h     = (const float*)d_in[3];
    const float* b_h     = (const float*)d_in[4];
    const float* W_c     = (const float*)d_in[5];
    const float* b_c     = (const float*)d_in[6];
    const float* W_ih    = (const float*)d_in[7];
    const float* W_hh    = (const float*)d_in[8];
    const float* b_ih    = (const float*)d_in[9];
    const float* b_hh    = (const float*)d_in[10];
    const float* W_out   = (const float*)d_in[11];
    const float* b_out   = (const float*)d_in[12];

    char* w = (char*)d_ws;
    size_t off = 0;
    auto alloc = [&](size_t bytes) { size_t cur = off; off = (off + bytes + 255) & ~(size_t)255; return cur; };

    float*  acc     = (float*) (w + alloc(2 * sizeof(float)));
    __bf16* src_bf  = (__bf16*)(w + alloc((size_t)BB * IMG * 2));
    __bf16* whc_bf  = (__bf16*)(w + alloc((size_t)2 * HDIM * IMG * 2));      // W_h rows then W_c rows
    __bf16* wih_bf  = (__bf16*)(w + alloc((size_t)4 * HDIM * IDIM * 2));
    __bf16* whh_bf  = (__bf16*)(w + alloc((size_t)4 * HDIM * HDIM * 2));
    __bf16* wout_bf = (__bf16*)(w + alloc((size_t)VOCAB * HDIM * 2));
    __bf16* x_bf    = (__bf16*)(w + alloc((size_t)BB * (TT - 1) * IDIM * 2));
    float*  c_state = (float*) (w + alloc((size_t)BB * HDIM * 4));
    __bf16* h_bf    = (__bf16*)(w + alloc((size_t)BB * HDIM * 2));
    float*  G       = (float*) (w + alloc((size_t)BB * 4 * HDIM * 4));
    __bf16* outs_bf = (__bf16*)(w + alloc((size_t)BB * (TT - 1) * HDIM * 2));
    float*  lse     = (float*) (w + alloc((size_t)NROWS * 4 + 256));         // tail slack

    zero_acc_kernel<<<1, 1, 0, stream>>>(acc);

    auto cvt = [&](const float* in, __bf16* out, int n) {
        convert_bf16_kernel<<<(n + 255) / 256, 256, 0, stream>>>(in, out, n);
    };
    cvt(src,   src_bf,                 BB * IMG);
    cvt(W_h,   whc_bf,                 HDIM * IMG);
    cvt(W_c,   whc_bf + (size_t)HDIM * IMG, HDIM * IMG);
    cvt(W_ih,  wih_bf,                 4 * HDIM * IDIM);
    cvt(W_hh,  whh_bf,                 4 * HDIM * HDIM);
    cvt(W_out, wout_bf,                VOCAB * HDIM);

    {
        int n = BB * (TT - 1) * IDIM;
        gather_x_kernel<<<(n + 255) / 256, 256, 0, stream>>>(embed, tgt_ids, x_bf, n);
    }

    init_hc_kernel<<<4 * 32, 32, 0, stream>>>(src_bf, whc_bf, b_h, b_c, c_state, h_bf);

    for (int t = 0; t < TT - 1; ++t) {
        gates_gemm_kernel<<<4 * 64, 32, 0, stream>>>(x_bf, h_bf, wih_bf, whh_bf, G, t);
        gate_update_kernel<<<(BB * HDIM + 255) / 256, 256, 0, stream>>>(G, b_ih, b_hh, c_state, h_bf, outs_bf, t);
    }

    lse_kernel<<<NROWS / 16, 256, 0, stream>>>(outs_bf, wout_bf, b_out, lse);
    loss_kernel<<<(NROWS + 7) / 8, 256, 0, stream>>>(outs_bf, W_out, b_out, lse, tgt_ids, acc);
    finalize_kernel<<<1, 1, 0, stream>>>(acc, (float*)d_out);
}
